// MAB_1726576855180
// MI455X (gfx1250) — matrix operand order repaired and checked by reading, not compiled, not run on hardware
//
#include <hip/hip_runtime.h>

typedef __attribute__((ext_vector_type(16))) __bf16 v16bf;
typedef __attribute__((ext_vector_type(8)))  __bf16 v8bf;
typedef __attribute__((ext_vector_type(8)))  float  v8f;

#define D_MODEL 1024
#define NHEAD   16
#define DHEAD   64
#define SEQ     2048
#define BATCH   4

__device__ __forceinline__ v8f zero_v8f() {
  v8f x;
#pragma unroll
  for (int i = 0; i < 8; ++i) x[i] = 0.0f;
  return x;
}

__device__ __forceinline__ v16bf cat16(v8bf lo, v8bf hi) {
  return __builtin_shufflevector(lo, hi, 0, 1, 2, 3, 4, 5, 6, 7,
                                 8, 9, 10, 11, 12, 13, 14, 15);
}

// ---------------- f32 -> bf16 conversion (linear) ----------------
__global__ void cvt_f32_bf16(const float* __restrict__ in, __bf16* __restrict__ out, int n) {
  int i = blockIdx.x * blockDim.x + threadIdx.x;
  int stride = gridDim.x * blockDim.x;
  for (; i < n; i += stride) out[i] = (__bf16)in[i];
}

// ---------------- f32 [R x C] -> bf16 transposed [C x R] ----------------
__global__ void cvt_f32_bf16_T(const float* __restrict__ in, __bf16* __restrict__ out,
                               int R, int C) {
  int i = blockIdx.x * blockDim.x + threadIdx.x;
  int total = R * C;
  int stride = gridDim.x * blockDim.x;
  for (; i < total; i += stride) {
    int r = i / C, c = i % C;
    out[(size_t)c * R + r] = (__bf16)in[i];
  }
}

// ---------------- tiled WMMA GEMM: out = epilogue(A[MxK] @ W + bias) ----------------
// W is passed TRANSPOSED: WT[N x K] so each lane's B-fragment is contiguous.
// block = 128 threads (4 waves); wave w computes rows [blk.x*64 + w*16, +16), cols [blk.y*64, +64)
__global__ __launch_bounds__(128)
void gemm_bf16_wmma(const __bf16* __restrict__ A, const __bf16* __restrict__ WT,
                    const float* __restrict__ bias, const float* __restrict__ resid,
                    float* __restrict__ outF, __bf16* __restrict__ outB,
                    __bf16* __restrict__ outBT,  // per-head transposed [B,H,DH,SEQ]
                    int M, int N, int K, int relu_resid) {
  const int lane = threadIdx.x & 31;
  const int wave = threadIdx.x >> 5;
  const int half = lane >> 4;
  const int l15  = lane & 15;
  const int row0 = blockIdx.x * 64 + wave * 16;
  const int col0 = blockIdx.y * 64;

  v8f acc[4];
#pragma unroll
  for (int t = 0; t < 4; ++t) acc[t] = zero_v8f();

  const __bf16* arow = A + (size_t)(row0 + l15) * K;

  for (int kk = 0; kk < K; kk += 32) {
    // A fragment: two contiguous 8-element chunks (16B vector loads)
    const v8bf* ap = reinterpret_cast<const v8bf*>(arow + kk);
    v16bf a = cat16(ap[2 * half], ap[2 * half + 1]);
#pragma unroll
    for (int t = 0; t < 4; ++t) {
      // B fragment: WT[col][kk + half*16 .. +15] contiguous (32B vector load)
      int col = col0 + t * 16 + l15;
      v16bf b = *reinterpret_cast<const v16bf*>(WT + (size_t)col * K + kk + half * 16);
      acc[t] = __builtin_amdgcn_wmma_f32_16x16x32_bf16(false, a, false, b,
                                                       (short)0, acc[t], false, false);
    }
  }

#pragma unroll
  for (int t = 0; t < 4; ++t) {
#pragma unroll
    for (int r = 0; r < 8; ++r) {
      int row = row0 + half * 8 + r;  // C/D layout: row = r + 8*half
      int col = col0 + t * 16 + l15;
      float v = acc[t][r] + bias[col];
      size_t idx = (size_t)row * N + col;
      if (relu_resid) v = resid[idx] + fmaxf(v, 0.0f);
      if (outF) outF[idx] = v;
      if (outB) outB[idx] = (__bf16)v;
      if (outBT) {
        int bb = row >> 11, s = row & (SEQ - 1);      // SEQ = 2048
        int hh = col >> 6,  d = col & (DHEAD - 1);    // DHEAD = 64
        outBT[(((size_t)bb * NHEAD + hh) * DHEAD + d) * SEQ + s] = (__bf16)v;
      }
    }
  }
}

// ---------------- flash-attention per (b,h,16-row q tile), 1 wave per tile ----------------
// Vt is per-head transposed: [B,H,DH,SEQ] so V fragments are contiguous over keys.
__global__ __launch_bounds__(128)
void attn_kernel(const __bf16* __restrict__ Qb, const __bf16* __restrict__ Kb,
                 const __bf16* __restrict__ Vt, const float* __restrict__ Qf,
                 float* __restrict__ O) {
  __shared__ __bf16 ldsP[4][16 * 32];
  const int lane = threadIdx.x & 31;
  const int wave = threadIdx.x >> 5;
  const int half = lane >> 4;
  const int l15  = lane & 15;
  const int b = blockIdx.z;
  const int h = blockIdx.y;
  const int q0 = (blockIdx.x * 4 + wave) * 16;

  const __bf16* qp  = Qb + (size_t)(b * SEQ + q0) * D_MODEL + h * DHEAD;
  const __bf16* kp  = Kb + (size_t)(b * SEQ) * D_MODEL + h * DHEAD;
  const __bf16* vpt = Vt + ((size_t)(b * NHEAD + h) * DHEAD) * SEQ;

  // q A-fragments for d in [0,32) and [32,64): 2x 16B vector loads each
  v16bf aq[2];
#pragma unroll
  for (int ds = 0; ds < 2; ++ds) {
    const v8bf* ap = reinterpret_cast<const v8bf*>(qp + (size_t)l15 * D_MODEL + ds * 32);
    aq[ds] = cat16(ap[2 * half], ap[2 * half + 1]);
  }

  float mrow[8], lrow[8];
#pragma unroll
  for (int r = 0; r < 8; ++r) { mrow[r] = -3.0e38f; lrow[r] = 0.0f; }
  v8f oacc[4];
#pragma unroll
  for (int t = 0; t < 4; ++t) oacc[t] = zero_v8f();

  const float scale = 0.03125f;  // 1/sqrt(D_MODEL)

  for (int key0 = 0; key0 < SEQ; key0 += 32) {
    // scores: 16 q rows x 32 keys (2 column tiles), contraction over dh=64 in 2 steps
    v8f s[2];
    s[0] = zero_v8f(); s[1] = zero_v8f();
#pragma unroll
    for (int ct = 0; ct < 2; ++ct) {
#pragma unroll
      for (int ds = 0; ds < 2; ++ds) {
        // B[k=d, n=key]: 16 contiguous d per lane (32B vector load)
        v16bf bk = *reinterpret_cast<const v16bf*>(
            kp + (size_t)(key0 + ct * 16 + l15) * D_MODEL + ds * 32 + half * 16);
        s[ct] = __builtin_amdgcn_wmma_f32_16x16x32_bf16(false, aq[ds], false, bk,
                                                        (short)0, s[ct], false, false);
      }
    }
    // online softmax: half-wave owns rows half*8 .. half*8+7, cols across 16 lanes
    float alpha[8];
#pragma unroll
    for (int r = 0; r < 8; ++r) {
      float x0 = s[0][r] * scale, x1 = s[1][r] * scale;
      float mx = fmaxf(x0, x1);
      mx = fmaxf(mx, __shfl_xor(mx, 1, 32));
      mx = fmaxf(mx, __shfl_xor(mx, 2, 32));
      mx = fmaxf(mx, __shfl_xor(mx, 4, 32));
      mx = fmaxf(mx, __shfl_xor(mx, 8, 32));
      float mn = fmaxf(mrow[r], mx);
      float al = __expf(mrow[r] - mn);
      float p0 = __expf(x0 - mn), p1 = __expf(x1 - mn);
      float sm = p0 + p1;
      sm += __shfl_xor(sm, 1, 32);
      sm += __shfl_xor(sm, 2, 32);
      sm += __shfl_xor(sm, 4, 32);
      sm += __shfl_xor(sm, 8, 32);
      lrow[r] = lrow[r] * al + sm;
      mrow[r] = mn;
      alpha[r] = al;
      // stage P (C/D layout -> LDS, row-major 16x32)
      ldsP[wave][(r + 8 * half) * 32 + l15]      = (__bf16)p0;
      ldsP[wave][(r + 8 * half) * 32 + 16 + l15] = (__bf16)p1;
    }
#pragma unroll
    for (int t = 0; t < 4; ++t)
#pragma unroll
      for (int r = 0; r < 8; ++r) oacc[t][r] *= alpha[r];

    asm volatile("" ::: "memory");  // keep DS store->load ordered (in-order per wave)
    // reload P in A-fragment layout: two 16B LDS vector loads
    const v8bf* lp = reinterpret_cast<const v8bf*>(&ldsP[wave][l15 * 32]);
    v16bf pa = cat16(lp[2 * half], lp[2 * half + 1]);
    asm volatile("" ::: "memory");

#pragma unroll
    for (int t = 0; t < 4; ++t) {
      // B[k=key, n=d] from transposed V: 16 contiguous keys per lane (32B load)
      v16bf bv = *reinterpret_cast<const v16bf*>(
          vpt + (size_t)(t * 16 + l15) * SEQ + key0 + half * 16);
      oacc[t] = __builtin_amdgcn_wmma_f32_16x16x32_bf16(false, pa, false, bv,
                                                        (short)0, oacc[t], false, false);
    }
  }

  // normalize + residual (+q)
  const float* qfp = Qf + (size_t)(b * SEQ + q0) * D_MODEL + h * DHEAD;
  float* op = O + (size_t)(b * SEQ + q0) * D_MODEL + h * DHEAD;
#pragma unroll
  for (int t = 0; t < 4; ++t)
#pragma unroll
    for (int r = 0; r < 8; ++r) {
      int row = half * 8 + r;
      int col = t * 16 + l15;
      float v = oacc[t][r] / lrow[r] + qfp[(size_t)row * D_MODEL + col];
      op[(size_t)row * D_MODEL + col] = v;
    }
}

// ---------------- layernorm over D=1024, one block per row ----------------
__global__ __launch_bounds__(256)
void layernorm_kernel(const float* __restrict__ X, const float* __restrict__ g,
                      const float* __restrict__ be, float* __restrict__ outF,
                      __bf16* __restrict__ outB) {
  __shared__ float red[256];
  const int row = blockIdx.x;
  const int t = threadIdx.x;
  const float* x = X + (size_t)row * D_MODEL;
  float v[4];
  float s = 0.0f;
#pragma unroll
  for (int i = 0; i < 4; ++i) { v[i] = x[t + i * 256]; s += v[i]; }
  red[t] = s;
  __syncthreads();
  for (int off = 128; off > 0; off >>= 1) {
    if (t < off) red[t] += red[t + off];
    __syncthreads();
  }
  float mean = red[0] * (1.0f / D_MODEL);
  __syncthreads();
  float s2 = 0.0f;
#pragma unroll
  for (int i = 0; i < 4; ++i) { float d = v[i] - mean; s2 += d * d; }
  red[t] = s2;
  __syncthreads();
  for (int off = 128; off > 0; off >>= 1) {
    if (t < off) red[t] += red[t + off];
    __syncthreads();
  }
  float rstd = rsqrtf(red[0] * (1.0f / D_MODEL) + 1e-5f);
#pragma unroll
  for (int i = 0; i < 4; ++i) {
    int c = t + i * 256;
    float y = (v[i] - mean) * rstd * g[c] + be[c];
    outF[(size_t)row * D_MODEL + c] = y;
    if (outB) outB[(size_t)row * D_MODEL + c] = (__bf16)y;
  }
}

extern "C" void kernel_launch(void* const* d_in, const int* in_sizes, int n_in,
                              void* d_out, int out_size, void* d_ws, size_t ws_size,
                              hipStream_t stream) {
  const float* Q  = (const float*)d_in[0];
  const float* K  = (const float*)d_in[1];
  const float* Wq = (const float*)d_in[2];
  const float* bq = (const float*)d_in[3];
  const float* Wk = (const float*)d_in[4];
  const float* bk = (const float*)d_in[5];
  const float* Wv = (const float*)d_in[6];
  const float* bv = (const float*)d_in[7];
  const float* Wo = (const float*)d_in[8];
  const float* bo = (const float*)d_in[9];
  const float* g0 = (const float*)d_in[10];
  const float* b0 = (const float*)d_in[11];
  const float* g1 = (const float*)d_in[12];
  const float* b1 = (const float*)d_in[13];
  float* out = (float*)d_out;

  char* ws = (char*)d_ws;
  const size_t MB = 1ull << 20;
  // workspace layout (regions reused once their producers/consumers are done):
  __bf16* Qb  = (__bf16*)(ws + 0);        // 16MB (dead after projections)
  __bf16* Kb  = (__bf16*)(ws + 16 * MB);  // 16MB (dead after projections... K stays for K-proj)
  float*  Ov  = (float*)(ws + 0);         // 32MB attention output, aliases Qb+Kb
  __bf16* Wqb = (__bf16*)(ws + 32 * MB);  // 2MB each, TRANSPOSED [N x K]
  __bf16* Wkb = (__bf16*)(ws + 34 * MB);
  __bf16* Wvb = (__bf16*)(ws + 36 * MB);
  __bf16* Wob = (__bf16*)(ws + 38 * MB);
  float*  qf  = (float*)(ws + 40 * MB);   // 32MB projected q (f32, residual)
  float*  t2  = qf;                       // FFN output aliases qf (dead by then)
  __bf16* qbb = (__bf16*)(ws + 72 * MB);  // 16MB projected q (bf16)
  __bf16* t1b = qbb;                      // post-LN0 bf16 aliases qbb
  __bf16* kbb = (__bf16*)(ws + 88 * MB);  // 16MB projected k (bf16, natural layout)
  __bf16* vbt = (__bf16*)(ws + 104 * MB); // 16MB projected v (bf16, [B,H,DH,SEQ])
  float*  t1  = (float*)(ws + 120 * MB);  // 32MB post-LN0 f32

  const int nQK = BATCH * SEQ * D_MODEL;

  cvt_f32_bf16<<<2048, 256, 0, stream>>>(Q, Qb, nQK);
  cvt_f32_bf16<<<2048, 256, 0, stream>>>(K, Kb, nQK);
  cvt_f32_bf16_T<<<1024, 256, 0, stream>>>(Wq, Wqb, D_MODEL, D_MODEL);
  cvt_f32_bf16_T<<<1024, 256, 0, stream>>>(Wk, Wkb, D_MODEL, D_MODEL);
  cvt_f32_bf16_T<<<1024, 256, 0, stream>>>(Wv, Wvb, D_MODEL, D_MODEL);
  cvt_f32_bf16_T<<<1024, 256, 0, stream>>>(Wo, Wob, D_MODEL, D_MODEL);

  dim3 ggrid((BATCH * SEQ) / 64, D_MODEL / 64);  // 128 x 16
  gemm_bf16_wmma<<<ggrid, 128, 0, stream>>>(Qb, Wqb, bq, nullptr, qf, qbb, nullptr,
                                            BATCH * SEQ, D_MODEL, D_MODEL, 0);
  gemm_bf16_wmma<<<ggrid, 128, 0, stream>>>(Kb, Wkb, bk, nullptr, nullptr, kbb, nullptr,
                                            BATCH * SEQ, D_MODEL, D_MODEL, 0);
  gemm_bf16_wmma<<<ggrid, 128, 0, stream>>>(Kb, Wvb, bv, nullptr, nullptr, nullptr, vbt,
                                            BATCH * SEQ, D_MODEL, D_MODEL, 0);

  dim3 agrid(SEQ / 64, NHEAD, BATCH);  // 32 x 16 x 4, 4 waves/block
  attn_kernel<<<agrid, 128, 0, stream>>>(qbb, kbb, vbt, qf, Ov);

  layernorm_kernel<<<BATCH * SEQ, 256, 0, stream>>>(Ov, g0, b0, t1, t1b);
  gemm_bf16_wmma<<<ggrid, 128, 0, stream>>>(t1b, Wob, bo, t1, t2, nullptr, nullptr,
                                            BATCH * SEQ, D_MODEL, D_MODEL, 1);
  layernorm_kernel<<<BATCH * SEQ, 256, 0, stream>>>(t2, g1, b1, out, nullptr);
}
